// SelfExcite2d_6803228197738
// MI455X (gfx1250) — compile-verified, matrix operand order checked by reading, not run on hardware
//
#include <hip/hip_runtime.h>
#include <hip/hip_bf16.h>

typedef __attribute__((ext_vector_type(16))) _Float16 v16h;
typedef __attribute__((ext_vector_type(8)))  _Float16 v8h;
typedef __attribute__((ext_vector_type(8)))  float    v8f;

#define BATCH 8
#define CDIM  512
#define HW    4096
#define DHEAD 64
#define EDIM  128   // 2*DHEAD

#if defined(__AMDGCN__) && __has_builtin(__builtin_amdgcn_global_load_async_to_lds_b128)
#define HAVE_ASYNC_LDS 1
typedef int v4i __attribute__((vector_size(16)));
typedef __attribute__((address_space(1))) v4i* gptr_v4i;   // global int4*
typedef __attribute__((address_space(3))) v4i* lptr_v4i;   // LDS int4*
#endif

// CDNA5 wave32 WMMA 16-bit A/B fragment: half h (0..15) of lane l holds
// K index k = (l&16 ? 8 : 0) + (h<8 ? h : h+8).  With K innermost in memory a
// fragment is two contiguous 16-byte loads at [row*K + hi] and [row*K + hi+16].
__device__ __forceinline__ v16h ldfrag(const _Float16* p, int hi) {
    v8h lo = *(const v8h*)(p + hi);
    v8h h8 = *(const v8h*)(p + hi + 16);
    return __builtin_shufflevector(lo, h8, 0,1,2,3,4,5,6,7,8,9,10,11,12,13,14,15);
}

__device__ __forceinline__ v16h ldfrag_scaled(const _Float16* p, int hi, _Float16 s) {
    v8h lo = *(const v8h*)(p + hi) * s;
    v8h h8 = *(const v8h*)(p + hi + 16) * s;
    return __builtin_shufflevector(lo, h8, 0,1,2,3,4,5,6,7,8,9,10,11,12,13,14,15);
}

__device__ __forceinline__ v8f wmma_f16(v16h a, v16h b, v8f c) {
    return __builtin_amdgcn_wmma_f32_16x16x32_f16(false, a, false, b, (short)0, c, false, false);
}

// -------- Kernel 0: f32 [rows][cols] -> f16 transposed [cols][rows] --------
__global__ void transpose_cvt_kernel(const float* __restrict__ src,
                                     _Float16* __restrict__ dst,
                                     int rows, int cols) {
    __shared__ float tile[16][17];
    const int tx = threadIdx.x & 15, ty = threadIdx.x >> 4;
    const int r0 = blockIdx.y * 16, c0 = blockIdx.x * 16;
    const size_t off = (size_t)blockIdx.z * rows * cols;
    tile[ty][tx] = src[off + (size_t)(r0 + ty) * cols + c0 + tx];
    __syncthreads();
    dst[off + (size_t)(c0 + ty) * rows + r0 + tx] = (_Float16)tile[tx][ty];
}

// -------- Kernel 1: tokens @ W_qk -> Qh, Kh  (f16, [B][HW][64]) ------------
// B fragments materialized as an array: 16 loads issue as one clause, the 8
// WMMAs drain with partial s_wait_loadcnt, live set ~150 VGPRs (no spills).
__global__ void qkproj_kernel(const _Float16* __restrict__ tok,   // [B][HW][C]
                              const _Float16* __restrict__ WqkT,  // [E][C]
                              const float* __restrict__ bqk,
                              _Float16* __restrict__ Qh,
                              _Float16* __restrict__ Kh) {
    const int lane = threadIdx.x;
    const int b  = blockIdx.y;
    const int t0 = blockIdx.x * 16;
    const int nl = lane & 15;
    const int hi = (lane & 16) ? 8 : 0;
    const _Float16* arow = tok + ((size_t)b * HW + t0 + nl) * CDIM;

    v8f acc[8];
    #pragma unroll
    for (int j = 0; j < 8; ++j) { v8f z = {}; acc[j] = z; }

    for (int c0 = 0; c0 < CDIM; c0 += 32) {
        v16h a = ldfrag(arow + c0, hi);
        v16h bf[8];
        #pragma unroll
        for (int j = 0; j < 8; ++j)
            bf[j] = ldfrag(WqkT + (size_t)(j * 16 + nl) * CDIM + c0, hi);
        #pragma unroll
        for (int j = 0; j < 8; ++j)
            acc[j] = wmma_f16(a, bf[j], acc[j]);
    }

    #pragma unroll
    for (int j = 0; j < 8; ++j) {
        int e = j * 16 + nl;
        float bias = bqk[e];
        #pragma unroll
        for (int i = 0; i < 8; ++i) {
            int t = t0 + i + hi;
            float v = acc[j][i] + bias;
            if (e < DHEAD)
                Qh[((size_t)b * HW + t) * DHEAD + e] = (_Float16)v;
            else
                Kh[((size_t)b * HW + t) * DHEAD + (e - DHEAD)] = (_Float16)v;
        }
    }
}

// -------- Kernel 2: V^T = W_v^T @ tokens^T  (f16, [B][C][HW]) --------------
__global__ void vproj_kernel(const _Float16* __restrict__ tok,  // [B][HW][C]
                             const _Float16* __restrict__ WvT,  // [C][C]
                             const float* __restrict__ bv,
                             _Float16* __restrict__ Vt) {
    const int lane = threadIdx.x;
    const int b  = blockIdx.z;
    const int e0 = blockIdx.y * 16;    // output channel tile (M)
    const int t0 = blockIdx.x * 128;   // token tile (N = 8 subtiles)
    const int nl = lane & 15;
    const int hi = (lane & 16) ? 8 : 0;
    const _Float16* arow = WvT + (size_t)(e0 + nl) * CDIM;
    const _Float16* brow = tok + ((size_t)b * HW + t0 + nl) * CDIM;

    v8f acc[8];
    #pragma unroll
    for (int j = 0; j < 8; ++j) { v8f z = {}; acc[j] = z; }

    for (int c0 = 0; c0 < CDIM; c0 += 32) {
        v16h a = ldfrag(arow + c0, hi);
        v16h bf[8];
        #pragma unroll
        for (int j = 0; j < 8; ++j)
            bf[j] = ldfrag(brow + (size_t)j * 16 * CDIM + c0, hi);
        #pragma unroll
        for (int j = 0; j < 8; ++j)
            acc[j] = wmma_f16(a, bf[j], acc[j]);
    }

    #pragma unroll
    for (int j = 0; j < 8; ++j) {
        #pragma unroll
        for (int i = 0; i < 8; ++i) {
            int e = e0 + i + hi;
            int t = t0 + j * 16 + nl;
            Vt[((size_t)b * CDIM + e) * HW + t] = (_Float16)(acc[j][i] + bv[e]);
        }
    }
}

// -------- Kernel 3: flash attention + residual -----------------------------
// One wave: 32 query rows x 128 output channels; double-buffered async V tile.
#if HAVE_ASYNC_LDS
__device__ __forceinline__ void issue_v_async(const _Float16* Vb, int cs, int kb,
                                              int lane, _Float16* dst) {
    const int sub = lane & 3, rb = lane >> 2;
    #pragma unroll
    for (int j = 0; j < 16; ++j) {
        int row = j * 8 + rb;
        const _Float16* g = Vb + (size_t)(cs * 128 + row) * HW + kb + sub * 8;
        __builtin_amdgcn_global_load_async_to_lds_b128(
            (gptr_v4i)(uintptr_t)g, (lptr_v4i)&dst[row * 32 + sub * 8], 0, 0);
    }
}
#endif

__global__ void attn_kernel(const _Float16* __restrict__ Qh,
                            const _Float16* __restrict__ Kh,
                            const _Float16* __restrict__ Vt,
                            const float* __restrict__ x,
                            const float* __restrict__ gamma,
                            float* __restrict__ out) {
    __shared__ __align__(16) _Float16 Plds[32 * 32];
#if HAVE_ASYNC_LDS
    __shared__ __align__(16) _Float16 Vlds[2][128 * 32];
#endif
    const int lane = threadIdx.x;
    const int b  = blockIdx.z;
    const int cs = blockIdx.y;          // 128-channel slice
    const int q0 = blockIdx.x * 32;     // 32-query tile
    const int nl = lane & 15;
    const int hi = (lane & 16) ? 8 : 0;

    const _Float16* Qb = Qh + (size_t)b * HW * DHEAD;
    const _Float16* Kb = Kh + (size_t)b * HW * DHEAD;
    const _Float16* Vb = Vt + (size_t)b * CDIM * HW;

    // Q fragments, pre-scaled by 1/sqrt(64)
    v16h qa[2][2];
    #pragma unroll
    for (int qs = 0; qs < 2; ++qs) {
        const _Float16* qrow = Qb + (size_t)(q0 + qs * 16 + nl) * DHEAD;
        qa[qs][0] = ldfrag_scaled(qrow, hi, (_Float16)0.125f);
        qa[qs][1] = ldfrag_scaled(qrow + 32, hi, (_Float16)0.125f);
    }

    float ms[2][8], ls[2][8];
    v8f acc[2][8];
    #pragma unroll
    for (int qs = 0; qs < 2; ++qs)
        #pragma unroll
        for (int i = 0; i < 8; ++i) { ms[qs][i] = -1e30f; ls[qs][i] = 0.f; }
    #pragma unroll
    for (int qs = 0; qs < 2; ++qs)
        #pragma unroll
        for (int t = 0; t < 8; ++t) { v8f z = {}; acc[qs][t] = z; }

#if HAVE_ASYNC_LDS
    issue_v_async(Vb, cs, 0, lane, &Vlds[0][0]);
#endif

    for (int kb = 0; kb < HW; kb += 32) {
        const int  cur  = (kb >> 5) & 1;
        const bool more = (kb + 32 < HW);
#if HAVE_ASYNC_LDS
        if (more) issue_v_async(Vb, cs, kb + 32, lane, &Vlds[cur ^ 1][0]);
#endif
        // K fragments for both 16-key subtiles (shared by both q subtiles)
        v16h kf[2][2];
        #pragma unroll
        for (int s = 0; s < 2; ++s) {
            const _Float16* krow = Kb + (size_t)(kb + s * 16 + nl) * DHEAD;
            kf[s][0] = ldfrag(krow, hi);
            kf[s][1] = ldfrag(krow + 32, hi);
        }
        #pragma unroll
        for (int qs = 0; qs < 2; ++qs) {
            v8f S0 = {}, S1 = {};
            S0 = wmma_f16(qa[qs][0], kf[0][0], S0);
            S0 = wmma_f16(qa[qs][1], kf[0][1], S0);
            S1 = wmma_f16(qa[qs][0], kf[1][0], S1);
            S1 = wmma_f16(qa[qs][1], kf[1][1], S1);
            // online softmax; row (qs*16 + i + hi) lives across a 16-lane half
            #pragma unroll
            for (int i = 0; i < 8; ++i) {
                float s0 = S0[i], s1 = S1[i];
                float mx = fmaxf(s0, s1);
                #pragma unroll
                for (int d = 1; d < 16; d <<= 1) mx = fmaxf(mx, __shfl_xor(mx, d, 32));
                float mn = fmaxf(ms[qs][i], mx);
                float alpha = __expf(ms[qs][i] - mn);
                ms[qs][i] = mn;
                float p0 = __expf(s0 - mn);
                float p1 = __expf(s1 - mn);
                float rs = p0 + p1;
                #pragma unroll
                for (int d = 1; d < 16; d <<= 1) rs += __shfl_xor(rs, d, 32);
                ls[qs][i] = ls[qs][i] * alpha + rs;
                #pragma unroll
                for (int t = 0; t < 8; ++t) acc[qs][t][i] *= alpha;
                Plds[(qs * 16 + i + hi) * 32 + nl]      = (_Float16)p0;
                Plds[(qs * 16 + i + hi) * 32 + 16 + nl] = (_Float16)p1;
            }
        }
#if HAVE_ASYNC_LDS
        // current batch of 16 asyncs done; next batch (16) may stay in flight
        if (more) __builtin_amdgcn_s_wait_asynccnt(16);
        else      __builtin_amdgcn_s_wait_asynccnt(0);
#endif
        __syncthreads();

        v16h pa[2];
        pa[0] = ldfrag(&Plds[(size_t)(nl) * 32], hi);
        pa[1] = ldfrag(&Plds[(size_t)(16 + nl) * 32], hi);

        #pragma unroll
        for (int t = 0; t < 8; ++t) {
#if HAVE_ASYNC_LDS
            v16h vf = ldfrag(&Vlds[cur][(size_t)(t * 16 + nl) * 32], hi);
#else
            v16h vf = ldfrag(Vb + (size_t)(cs * 128 + t * 16 + nl) * HW + kb, hi);
#endif
            acc[0][t] = wmma_f16(pa[0], vf, acc[0][t]);
            acc[1][t] = wmma_f16(pa[1], vf, acc[1][t]);
        }
        __syncthreads();
    }

    // epilogue: out = x + gamma * (acc / l)
    const float g = gamma[0];
    #pragma unroll
    for (int qs = 0; qs < 2; ++qs)
        #pragma unroll
        for (int t = 0; t < 8; ++t) {
            int c = cs * 128 + t * 16 + nl;
            #pragma unroll
            for (int i = 0; i < 8; ++i) {
                int tok = q0 + qs * 16 + i + hi;
                size_t idx = ((size_t)b * CDIM + c) * HW + tok;
                out[idx] = x[idx] + g * (acc[qs][t][i] / ls[qs][i]);
            }
        }
}

extern "C" void kernel_launch(void* const* d_in, const int* in_sizes, int n_in,
                              void* d_out, int out_size, void* d_ws, size_t ws_size,
                              hipStream_t stream) {
    const float* x     = (const float*)d_in[0];
    const float* Wqk   = (const float*)d_in[1];
    const float* bqk   = (const float*)d_in[2];
    const float* Wv    = (const float*)d_in[3];
    const float* bv    = (const float*)d_in[4];
    const float* gamma = (const float*)d_in[5];
    float* out = (float*)d_out;

    _Float16* Qh   = (_Float16*)d_ws;                          //  4 MiB
    _Float16* Kh   = Qh  + (size_t)BATCH * HW * DHEAD;         //  4 MiB
    _Float16* Vt   = Kh  + (size_t)BATCH * HW * DHEAD;         // 32 MiB
    _Float16* tokh = Vt  + (size_t)BATCH * CDIM * HW;          // 32 MiB
    _Float16* WqkT = tokh + (size_t)BATCH * HW * CDIM;         // 128 KiB
    _Float16* WvT  = WqkT + (size_t)EDIM * CDIM;               // 512 KiB

    // one-time f16 staging (K-innermost layouts for b128 fragment loads)
    transpose_cvt_kernel<<<dim3(HW / 16, CDIM / 16, BATCH), 256, 0, stream>>>(x, tokh, CDIM, HW);
    transpose_cvt_kernel<<<dim3(EDIM / 16, CDIM / 16, 1), 256, 0, stream>>>(Wqk, WqkT, CDIM, EDIM);
    transpose_cvt_kernel<<<dim3(CDIM / 16, CDIM / 16, 1), 256, 0, stream>>>(Wv, WvT, CDIM, CDIM);

    qkproj_kernel<<<dim3(HW / 16, BATCH), 32, 0, stream>>>(tokh, WqkT, bqk, Qh, Kh);
    vproj_kernel<<<dim3(HW / 128, CDIM / 16, BATCH), 32, 0, stream>>>(tokh, WvT, bv, Vt);
    attn_kernel<<<dim3(HW / 32, CDIM / 128, BATCH), 32, 0, stream>>>(Qh, Kh, Vt, x, gamma, out);
}
